// BBoxHeadForGroundTruthBboxRegressionV2_25228637897253
// MI455X (gfx1250) — compile-verified, matrix operand order checked by reading, not run on hardware
//
#include <hip/hip_runtime.h>
#include <hip/hip_bf16.h>

typedef __attribute__((ext_vector_type(16))) _Float16 v16h;
typedef __attribute__((ext_vector_type(8)))  _Float16 v8h;
typedef __attribute__((ext_vector_type(8)))  float    v8f;

#define TILE_ROWS 128
#define TILE_COLS 64
#define TILE_K    32
#define LDS_STRIDE 40   // 32 + 8 halves padding (80B row pitch)

// ---------------------------------------------------------------------------
// f32 -> f16 conversion (vectorized, 8 elements/thread)
// ---------------------------------------------------------------------------
__global__ void __launch_bounds__(256)
f32_to_f16_k(const float* __restrict__ in, _Float16* __restrict__ out, long long n)
{
    long long i = ((long long)blockIdx.x * 256 + threadIdx.x) * 8;
    if (i + 8 <= n) {
        float4 a = *(const float4*)(in + i);
        float4 b = *(const float4*)(in + i + 4);
        v8h o;
        o[0] = (_Float16)a.x; o[1] = (_Float16)a.y;
        o[2] = (_Float16)a.z; o[3] = (_Float16)a.w;
        o[4] = (_Float16)b.x; o[5] = (_Float16)b.y;
        o[6] = (_Float16)b.z; o[7] = (_Float16)b.w;
        *(v8h*)(out + i) = o;
    } else {
        for (long long j = i; j < n; ++j) out[j] = (_Float16)in[j];
    }
}

// ---------------------------------------------------------------------------
// Async global->LDS copy of one 16-byte chunk (8 halves). ASYNCcnt-tracked.
// LDS address = low 32 bits of generic pointer (ISA aperture rule).
// ---------------------------------------------------------------------------
__device__ __forceinline__ void async_copy_b128(const _Float16* gsrc, unsigned lds_off)
{
    unsigned long long gaddr = (unsigned long long)gsrc;
    asm volatile("global_load_async_to_lds_b128 %0, %1, off"
                 :: "v"(lds_off), "v"(gaddr) : "memory");
}
__device__ __forceinline__ unsigned lds_addr(const void* p)
{
    return (unsigned)(unsigned long long)p;
}
__device__ __forceinline__ void wait_async_le(int n)   // n is compile-time 0 or 3
{
    if (n == 0) asm volatile("s_wait_asynccnt 0x0" ::: "memory");
    else        asm volatile("s_wait_asynccnt 0x3" ::: "memory");
}

// ---------------------------------------------------------------------------
// GEMM: C[nrows, M] = act(A[nrows,K] @ W[M,K]^T + bias) (+ residual, f32)
// A, W f16; f32 accumulate via v_wmma_f32_16x16x32_f16; C is f32 or f16.
// Requirements: nrows % 128 == 0 (all call sites), K % 32 == 0.
// Ragged M handled by source-row clamp + uniform-branch guarded epilogue.
// Double-buffered LDS; tensor DMA for tile k+1 in flight during tile k's WMMA
// (ASYNCcnt decrements in order; 3 async ops per tile per wave).
// ---------------------------------------------------------------------------
template <typename OT, int ACT, int HAS_RES>
__global__ void __launch_bounds__(256)
gemm_bias_act(const _Float16* __restrict__ A, const _Float16* __restrict__ W,
              const float* __restrict__ bias, const float* __restrict__ R,
              OT* __restrict__ C, int nrows, int M, int K, int ldc)
{
    __shared__ _Float16 Ash[2][TILE_ROWS * LDS_STRIDE];
    __shared__ _Float16 Wsh[2][TILE_COLS * LDS_STRIDE];

    const int tid  = threadIdx.x;
    const int lane = tid & 31;
    const int wave = tid >> 5;
    const int wm   = wave & 3;   // 4 row-waves
    const int wn   = wave >> 2;  // 2 col-waves
    const int rowBase = blockIdx.y * TILE_ROWS;
    const int colBase = blockIdx.x * TILE_COLS;

    // ---- per-thread staging state: chunk = 8 halves (16B) -----------------
    const int ko = (tid & 3) * 8;      // k-offset within tile
    const int rr = tid >> 2;           // 0..63
    int ga0 = rowBase + rr;       if (ga0 > nrows - 1) ga0 = nrows - 1;
    int ga1 = rowBase + rr + 64;  if (ga1 > nrows - 1) ga1 = nrows - 1;
    int gw  = colBase + rr;       if (gw  > M - 1)     gw  = M - 1;
    const _Float16* pa0 = A + (size_t)ga0 * K + ko;
    const _Float16* pa1 = A + (size_t)ga1 * K + ko;
    const _Float16* pw  = W + (size_t)gw  * K + ko;
    unsigned la0[2], la1[2], lw0[2];
    #pragma unroll
    for (int b = 0; b < 2; ++b) {
        la0[b] = lds_addr(&Ash[b][rr * LDS_STRIDE + ko]);
        la1[b] = la0[b] + 64 * LDS_STRIDE * 2;
        lw0[b] = lds_addr(&Wsh[b][rr * LDS_STRIDE + ko]);
    }

    auto issue_tile = [&](int kt, int b) {
        const _Float16* s0 = pa0 + (size_t)kt * TILE_K;
        const _Float16* s1 = pa1 + (size_t)kt * TILE_K;
        const _Float16* sw = pw  + (size_t)kt * TILE_K;
        async_copy_b128(s0, la0[b]);
        async_copy_b128(s1, la1[b]);
        async_copy_b128(sw, lw0[b]);
    };

    v8f acc[2][2] = {};

    const int nk = K / TILE_K;
    issue_tile(0, 0);
    if (nk > 1) issue_tile(1, 1);

    const int kb   = (lane < 16) ? 0 : 8;   // ISA 16-bit A-frag layout
    const int rsel = lane & 15;

    for (int kk = 0; kk < nk; ++kk) {
        const int b = kk & 1;
        if (kk + 1 < nk) wait_async_le(3);  // tile kk landed; kk+1 may fly
        else             wait_async_le(0);
        __syncthreads();                    // all waves' DMA for tile kk visible

        const _Float16* As = Ash[b];
        const _Float16* Ws = Wsh[b];
        v16h af[2], bf[2];
        #pragma unroll
        for (int f = 0; f < 2; ++f) {
            const _Float16* ap = &As[(wm * 32 + f * 16 + rsel) * LDS_STRIDE + kb];
            const _Float16* bp = &Ws[(wn * 32 + f * 16 + rsel) * LDS_STRIDE + kb];
            #pragma unroll
            for (int e = 0; e < 8; ++e) {
                af[f][e]     = ap[e];
                af[f][e + 8] = ap[16 + e];
                bf[f][e]     = bp[e];
                bf[f][e + 8] = bp[16 + e];
            }
        }
        #pragma unroll
        for (int fm = 0; fm < 2; ++fm)
            #pragma unroll
            for (int fn = 0; fn < 2; ++fn)
                acc[fm][fn] = __builtin_amdgcn_wmma_f32_16x16x32_f16(
                    false, af[fm], false, bf[fn], (short)0, acc[fm][fn], false, false);

        if (kk + 1 < nk) {
            __syncthreads();                // all waves done reading buffer b
            if (kk + 2 < nk) issue_tile(kk + 2, b);   // overwrite b for tile kk+2
        }
    }

    // ---- epilogue: C/D layout — VGPR r: lanes 0-15 -> M=r, N=lane;
    //                lanes 16-31 -> M=r+8, N=lane-16 --------------------------
    const int rhalf = (lane >> 4) * 8;
    const int ncol  = lane & 15;
    const int rb    = rowBase + wm * 32;
    const int cb    = colBase + wn * 32;

    if (colBase + TILE_COLS <= M) {          // uniform fast path, no guards
        #pragma unroll
        for (int fn = 0; fn < 2; ++fn) {
            const int col = cb + fn * 16 + ncol;
            const float bv = bias[col];
            #pragma unroll
            for (int fm = 0; fm < 2; ++fm)
                #pragma unroll
                for (int r = 0; r < 8; ++r) {
                    const int row = rb + fm * 16 + rhalf + r;
                    float v = acc[fm][fn][r] + bv;
                    if (ACT) v = fmaxf(v, 0.0f);
                    if (HAS_RES) v += R[(size_t)row * ldc + col];
                    C[(size_t)row * ldc + col] = (OT)v;
                }
        }
    } else {                                  // ragged M (final 6-col GEMM only)
        #pragma unroll
        for (int fn = 0; fn < 2; ++fn) {
            const int col = cb + fn * 16 + ncol;
            if (col < M) {
                const float bv = bias[col];
                #pragma unroll
                for (int fm = 0; fm < 2; ++fm)
                    #pragma unroll
                    for (int r = 0; r < 8; ++r) {
                        const int row = rb + fm * 16 + rhalf + r;
                        float v = acc[fm][fn][r] + bv;
                        if (ACT) v = fmaxf(v, 0.0f);
                        if (HAS_RES) v += R[(size_t)row * ldc + col];
                        C[(size_t)row * ldc + col] = (OT)v;
                    }
            }
        }
    }
}

// ---------------------------------------------------------------------------
// LayerNorm over D=1024; one wave per row; f32 in, f16 out (feeds GEMM only).
// ---------------------------------------------------------------------------
__global__ void __launch_bounds__(256)
layernorm_k(const float* __restrict__ X, _Float16* __restrict__ Y,
            const float* __restrict__ g, const float* __restrict__ b,
            int nrows, long long in_stride, long long out_stride)
{
    const int lane = threadIdx.x & 31;
    const int wave = threadIdx.x >> 5;
    const int row  = blockIdx.x * 8 + wave;
    if (row >= nrows) return;
    const float* xr = X + (size_t)row * in_stride;
    float v[32];
    float s = 0.0f;
    #pragma unroll
    for (int i = 0; i < 32; ++i) { v[i] = xr[lane + 32 * i]; s += v[i]; }
    #pragma unroll
    for (int off = 16; off; off >>= 1) s += __shfl_xor(s, off, 32);
    const float mu = s * (1.0f / 1024.0f);
    float q = 0.0f;
    #pragma unroll
    for (int i = 0; i < 32; ++i) { float d = v[i] - mu; q += d * d; }
    #pragma unroll
    for (int off = 16; off; off >>= 1) q += __shfl_xor(q, off, 32);
    const float rstd = rsqrtf(q * (1.0f / 1024.0f) + 1e-5f);
    _Float16* yr = Y + (size_t)row * out_stride;
    #pragma unroll
    for (int i = 0; i < 32; ++i) {
        int c = lane + 32 * i;
        yr[c] = (_Float16)((v[i] - mu) * rstd * g[c] + b[c]);
    }
}

// ---------------------------------------------------------------------------
// Build x[t, s, :]: s==0 -> lang[t], s>=1 -> visproj[batch_idx[t], s-1]
// ---------------------------------------------------------------------------
__global__ void __launch_bounds__(256)
build_x_k(const float* __restrict__ lang, const float* __restrict__ visproj,
          const int* __restrict__ bidx, float* __restrict__ x)
{
    size_t i = (size_t)blockIdx.x * 256 + threadIdx.x;   // total = 256*257*1024
    int col = (int)(i & 1023);
    size_t row = i >> 10;
    int s = (int)(row % 257);
    int t = (int)(row / 257);
    float v;
    if (s == 0) v = lang[(size_t)t * 1024 + col];
    else        v = visproj[((size_t)bidx[t] * 256 + (s - 1)) * 1024 + col];
    x[i] = v;
}

// ---------------------------------------------------------------------------
// Attention: one block per (t,h); 8 waves, one query per wave at a time.
// S=257, dh=256, qkv row stride 3072. qkv f32 in, o f16 out (feeds Wo GEMM).
// ---------------------------------------------------------------------------
__global__ void __launch_bounds__(256)
attention_k(const float* __restrict__ qkv, _Float16* __restrict__ o)
{
    __shared__ float wts[8][264];
    const int lane = threadIdx.x & 31;
    const int wave = threadIdx.x >> 5;
    const int t = blockIdx.x >> 2;
    const int h = blockIdx.x & 3;
    const float scale = 0.0625f; // 1/sqrt(256)
    const size_t base = (size_t)t * 257 * 3072 + (size_t)h * 256;
    const float* qb = qkv + base;
    const float* kb = qkv + base + 1024;
    const float* vb = qkv + base + 2048;

    for (int q = wave; q < 257; q += 8) {
        const float* qr = qb + (size_t)q * 3072;
        float qv[8];
        #pragma unroll
        for (int i = 0; i < 8; ++i) qv[i] = qr[lane + 32 * i];

        float smax = -1e30f;
        for (int j = 0; j < 257; ++j) {
            const float* kr = kb + (size_t)j * 3072;
            float p = 0.0f;
            #pragma unroll
            for (int i = 0; i < 8; ++i) p += qv[i] * kr[lane + 32 * i];
            #pragma unroll
            for (int off = 16; off; off >>= 1) p += __shfl_xor(p, off, 32);
            p *= scale;
            smax = fmaxf(smax, p);
            if (lane == 0) wts[wave][j] = p;
        }
        float lsum = 0.0f;
        for (int j = lane; j < 257; j += 32) {
            float e = __expf(wts[wave][j] - smax);
            wts[wave][j] = e;
            lsum += e;
        }
        #pragma unroll
        for (int off = 16; off; off >>= 1) lsum += __shfl_xor(lsum, off, 32);
        const float inv = 1.0f / lsum;

        float acc[8] = {0, 0, 0, 0, 0, 0, 0, 0};
        for (int j = 0; j < 257; ++j) {
            float w = wts[wave][j] * inv;
            const float* vr = vb + (size_t)j * 3072;
            #pragma unroll
            for (int i = 0; i < 8; ++i) acc[i] += w * vr[lane + 32 * i];
        }
        _Float16* op = o + ((size_t)t * 257 + q) * 1024 + (size_t)h * 256;
        #pragma unroll
        for (int i = 0; i < 8; ++i) op[lane + 32 * i] = (_Float16)acc[i];
    }
}

// ---------------------------------------------------------------------------
static inline dim3 gemm_grid(int nrows, int M)
{
    return dim3((M + TILE_COLS - 1) / TILE_COLS, (nrows + TILE_ROWS - 1) / TILE_ROWS);
}

static inline void gemm16_relu(hipStream_t s, const _Float16* A, const _Float16* W,
                               const float* bias, _Float16* C, int nrows, int M, int K, int ldc)
{
    gemm_bias_act<_Float16, 1, 0><<<gemm_grid(nrows, M), dim3(256), 0, s>>>(A, W, bias, nullptr, C, nrows, M, K, ldc);
}
static inline void gemm32_plain(hipStream_t s, const _Float16* A, const _Float16* W,
                                const float* bias, float* C, int nrows, int M, int K, int ldc)
{
    gemm_bias_act<float, 0, 0><<<gemm_grid(nrows, M), dim3(256), 0, s>>>(A, W, bias, nullptr, C, nrows, M, K, ldc);
}
static inline void gemm32_res(hipStream_t s, const _Float16* A, const _Float16* W,
                              const float* bias, const float* R, float* C, int nrows, int M, int K, int ldc)
{
    gemm_bias_act<float, 0, 1><<<gemm_grid(nrows, M), dim3(256), 0, s>>>(A, W, bias, R, C, nrows, M, K, ldc);
}

static inline void lnorm(hipStream_t s, const float* X, _Float16* Y,
                         const float* g, const float* b, int nrows,
                         long long istride, long long ostride)
{
    layernorm_k<<<dim3((nrows + 7) / 8), dim3(256), 0, s>>>(X, Y, g, b, nrows, istride, ostride);
}

static inline void conv16(hipStream_t s, const float* in, _Float16* out, long long n)
{
    unsigned blocks = (unsigned)((n + 2047) / 2048);
    f32_to_f16_k<<<dim3(blocks), dim3(256), 0, s>>>(in, out, n);
}

extern "C" void kernel_launch(void* const* d_in, const int* in_sizes, int n_in,
                              void* d_out, int out_size, void* d_ws, size_t ws_size,
                              hipStream_t stream)
{
    (void)in_sizes; (void)n_in; (void)out_size; (void)ws_size;

    const float* grd    = (const float*)d_in[0];   // (256, 4096)
    const int*   bidx   = (const int*)  d_in[1];   // (256,)
    const float* vision = (const float*)d_in[2];   // (16, 256, 1024)
    const float* vis_W  = (const float*)d_in[3];
    const float* vis_b  = (const float*)d_in[4];
    const float* lang_W1= (const float*)d_in[5];
    const float* lang_b1= (const float*)d_in[6];
    const float* lang_W2= (const float*)d_in[7];
    const float* lang_b2= (const float*)d_in[8];
    const float* lang_W3= (const float*)d_in[9];
    const float* lang_b3= (const float*)d_in[10];
    const float* Wqkv   = (const float*)d_in[11];  // (4, 3072, 1024)
    const float* bqkv   = (const float*)d_in[12];
    const float* Wo     = (const float*)d_in[13];  // (4, 1024, 1024)
    const float* bo     = (const float*)d_in[14];
    const float* ln1g   = (const float*)d_in[15];
    const float* ln1b   = (const float*)d_in[16];
    const float* ln2g   = (const float*)d_in[17];
    const float* ln2b   = (const float*)d_in[18];
    const float* Wff1   = (const float*)d_in[19];
    const float* bff1   = (const float*)d_in[20];
    const float* Wff2   = (const float*)d_in[21];
    const float* bff2   = (const float*)d_in[22];
    const float* lnfg   = (const float*)d_in[23];
    const float* lnfb   = (const float*)d_in[24];
    const float* box_W1 = (const float*)d_in[25];
    const float* box_b1 = (const float*)d_in[26];
    const float* box_W2 = (const float*)d_in[27];
    const float* box_b2 = (const float*)d_in[28];
    const float* box_W3 = (const float*)d_in[29];
    const float* box_b3 = (const float*)d_in[30];
    const float* box_W4 = (const float*)d_in[31];
    const float* box_b4 = (const float*)d_in[32];
    const float* box_W5 = (const float*)d_in[33];  // (6, 1024)
    const float* box_b5 = (const float*)d_in[34];

    const int T = 256, S = 257, D = 1024, BL = 16 * 256;
    const int ROWS = T * S;  // 65792 = 514 * 128

    // -------- workspace layout: f32 region first, then f16 region ----------
    float* fp = (float*)d_ws;
    float* visproj = fp; fp += (size_t)BL * D;          // f32 (feeds build_x)
    float* langOut = fp; fp += (size_t)T * D;           // f32 (feeds build_x)
    float* x       = fp; fp += (size_t)ROWS * D;        // f32 residual stream
    float* qkv     = fp; fp += (size_t)ROWS * 3 * D;    // f32 (attention input)

    _Float16* hp = (_Float16*)fp;
    _Float16* grd16    = hp; hp += (size_t)T * 4096;
    _Float16* vis16    = hp; hp += (size_t)BL * D;
    _Float16* visW16   = hp; hp += (size_t)1024 * 1024;
    _Float16* lW1_16   = hp; hp += (size_t)1024 * 4096;
    _Float16* lW2_16   = hp; hp += (size_t)1024 * 1024;
    _Float16* lW3_16   = hp; hp += (size_t)1024 * 1024;
    _Float16* Wqkv16   = hp; hp += (size_t)4 * 3072 * 1024;
    _Float16* Wo16     = hp; hp += (size_t)4 * 1024 * 1024;
    _Float16* Wff1_16  = hp; hp += (size_t)4 * 1024 * 1024;
    _Float16* Wff2_16  = hp; hp += (size_t)4 * 1024 * 1024;
    _Float16* bW16_1   = hp; hp += (size_t)1024 * 1024;
    _Float16* bW16_2   = hp; hp += (size_t)1024 * 1024;
    _Float16* bW16_3   = hp; hp += (size_t)1024 * 1024;
    _Float16* bW16_4   = hp; hp += (size_t)1024 * 1024;
    _Float16* bW16_5   = hp; hp += (size_t)8 * 1024;      // 6x1024 padded
    _Float16* langA16  = hp; hp += (size_t)T * D;
    _Float16* langB16  = hp; hp += (size_t)T * D;
    _Float16* x2_16    = hp; hp += (size_t)ROWS * D;
    _Float16* o16      = hp; hp += (size_t)ROWS * D;
    _Float16* fused16  = hp; hp += (size_t)T * D;
    _Float16* bh1_16   = hp; hp += (size_t)T * D;
    _Float16* bh2_16   = hp; hp += (size_t)T * D;
    _Float16* ffh16    = (_Float16*)qkv;                 // alias: qkv dead after attention

    // -------- 0) one-time f32 -> f16 conversions ---------------------------
    conv16(stream, grd,    grd16,   (long long)T * 4096);
    conv16(stream, vision, vis16,   (long long)BL * D);
    conv16(stream, vis_W,  visW16,  1024LL * 1024);
    conv16(stream, lang_W1,lW1_16,  1024LL * 4096);
    conv16(stream, lang_W2,lW2_16,  1024LL * 1024);
    conv16(stream, lang_W3,lW3_16,  1024LL * 1024);
    conv16(stream, Wqkv,   Wqkv16,  4LL * 3072 * 1024);
    conv16(stream, Wo,     Wo16,    4LL * 1024 * 1024);
    conv16(stream, Wff1,   Wff1_16, 4LL * 1024 * 1024);
    conv16(stream, Wff2,   Wff2_16, 4LL * 1024 * 1024);
    conv16(stream, box_W1, bW16_1,  1024LL * 1024);
    conv16(stream, box_W2, bW16_2,  1024LL * 1024);
    conv16(stream, box_W3, bW16_3,  1024LL * 1024);
    conv16(stream, box_W4, bW16_4,  1024LL * 1024);
    conv16(stream, box_W5, bW16_5,  6LL * 1024);

    // -------- 1) vision projection (f32 out for build_x) -------------------
    gemm32_plain(stream, vis16, visW16, vis_b, visproj, BL, D, 1024, D);

    // -------- 2) language MLP ----------------------------------------------
    gemm16_relu(stream, grd16,   lW1_16, lang_b1, langA16, T, 1024, 4096, 1024);
    gemm16_relu(stream, langA16, lW2_16, lang_b2, langB16, T, 1024, 1024, 1024);
    gemm32_plain(stream, langB16, lW3_16, lang_b3, langOut, T, D, 1024, D);

    // -------- 3) concat lang token + gathered vision -> x (f32) ------------
    {
        size_t total = (size_t)ROWS * D;
        build_x_k<<<dim3((unsigned)(total / 256)), dim3(256), 0, stream>>>(langOut, visproj, bidx, x);
    }

    // -------- 4) encoder layers --------------------------------------------
    for (int i = 0; i < 4; ++i) {
        const size_t wOff = (size_t)i * 1024 * 1024;
        lnorm(stream, x, x2_16, ln1g + i * D, ln1b + i * D, ROWS, D, D);
        gemm32_plain(stream, x2_16, Wqkv16 + (size_t)i * 3072 * 1024, bqkv + i * 3072,
                     qkv, ROWS, 3 * D, D, 3 * D);
        attention_k<<<dim3(T * 4), dim3(256), 0, stream>>>(qkv, o16);
        gemm32_res(stream, o16, Wo16 + wOff, bo + i * D, x, x, ROWS, D, D, D);
        lnorm(stream, x, x2_16, ln2g + i * D, ln2b + i * D, ROWS, D, D);
        gemm16_relu(stream, x2_16, Wff1_16 + wOff, bff1 + i * D, ffh16, ROWS, D, D, D);
        gemm32_res(stream, ffh16, Wff2_16 + wOff, bff2 + i * D, x, x, ROWS, D, D, D);
    }

    // -------- 5) final LN on s==0 rows -> fused (f16) ----------------------
    lnorm(stream, x, fused16, lnfg, lnfb, T, (long long)S * D, D);

    // -------- 6) box head ---------------------------------------------------
    gemm16_relu(stream, fused16, bW16_1, box_b1, bh1_16, T, 1024, 1024, 1024);
    gemm16_relu(stream, bh1_16,  bW16_2, box_b2, bh2_16, T, 1024, 1024, 1024);
    gemm16_relu(stream, bh2_16,  bW16_3, box_b3, bh1_16, T, 1024, 1024, 1024);
    gemm16_relu(stream, bh1_16,  bW16_4, box_b4, bh2_16, T, 1024, 1024, 1024);
    gemm32_plain(stream, bh2_16, bW16_5, box_b5, (float*)d_out, T, 6, 1024, 6);
}
